// SC2AtariNetActorCritic_11295763988523
// MI455X (gfx1250) — matrix operand order repaired and checked by reading, not compiled, not run on hardware
//
#include <hip/hip_runtime.h>

// ---------------------------------------------------------------------------
// CDNA5 (gfx1250) wave32 WMMA implementation of the SC2/Atari relational net.
// All matmul-shaped work runs through v_wmma_f32_16x16x32_f16 (f16 in, f32 acc).
// K-loops are software-pipelined in registers: next-step fragments are loaded
// before the current step's WMMAs so load latency overlaps the matrix pipe.
// ---------------------------------------------------------------------------

typedef __attribute__((ext_vector_type(16))) _Float16 v16h;
typedef __attribute__((ext_vector_type(8)))  _Float16 v8h;
typedef __attribute__((ext_vector_type(8)))  float    v8f;

#define WAVE 32

__device__ __forceinline__ v8f wmma16(v16h a, v16h b, v8f c) {
  // D = A(16x32) * B(32x16) + C, f32 accumulate
  return __builtin_amdgcn_wmma_f32_16x16x32_f16(
      /*neg_a=*/false, a, /*neg_b=*/false, b,
      /*c_mod=*/(short)0, c, /*reuse_a=*/false, /*reuse_b=*/false);
}

// Assemble the A fragment from two contiguous 16B chunks (K 0-15 of
// this lane's half), branchless.
__device__ __forceinline__ v16h make_a(const _Float16* p) {
  v8h lo = *(const v8h*)(p);        // K = k0 .. k0+7
  v8h hi = *(const v8h*)(p + 8);    // K = k0+8 .. k0+15
  v16h a;
#pragma unroll
  for (int j = 0; j < 8; ++j) { a[j] = lo[j]; a[j + 8] = hi[j]; }
  return a;
}

// --- A fragment, fully predicated scalar path (ragged M/K, any lda).
__device__ __forceinline__ v16h load_a_pred(const _Float16* A, int lda,
                                            int mbase, int kbase, int lane,
                                            int mlim, int klim) {
  const int m  = mbase + (lane & 15);
  const int k0 = kbase + ((lane >> 4) << 4);
  v16h a = {};
  if (m < mlim) {
    const _Float16* p = A + (size_t)m * lda;
#pragma unroll
    for (int j = 0; j < 16; ++j) {
      const int k = k0 + j;
      if (k < klim) a[j] = p[k];
    }
  }
  return a;
}

// --- B fragment from W stored row-major [N][K] (C = A * W^T), predicated.
__device__ __forceinline__ v16h load_bT_pred(const _Float16* W, int ldw,
                                             int nbase, int kbase, int lane,
                                             int nlim) {
  const int n  = nbase + (lane & 15);
  const int k0 = kbase + ((lane >> 4) << 4);
  v16h b = {};
  if (n < nlim) {
    const _Float16* p = W + (size_t)n * ldw + k0;
#pragma unroll
    for (int j = 0; j < 16; ++j) b[j] = p[j];
  }
  return b;
}

// --- B fragment from B stored row-major [K][N] (direct), ragged K allowed.
__device__ __forceinline__ v16h load_b_direct(const _Float16* Bm, int ldb,
                                              int nbase, int kbase, int lane,
                                              int klim) {
  const int n  = nbase + (lane & 15);
  const int k0 = kbase + ((lane >> 4) << 4);
  v16h b = {};
#pragma unroll
  for (int j = 0; j < 16; ++j) {
    const int k = k0 + j;
    if (k < klim) b[j] = Bm[(size_t)k * ldb + n];
  }
  return b;
}

// ---------------------------------------------------------------------------
// conv1: direct f32 conv, 17ch 8x8 stride4, lrelu, f16 output [256][64][20][20]
// ---------------------------------------------------------------------------
__global__ void k_conv1(const float* __restrict__ x, const float* __restrict__ w,
                        const float* __restrict__ bias, _Float16* __restrict__ y) {
  int idx = blockIdx.x * blockDim.x + threadIdx.x;
  const int total = 256 * 64 * 20 * 20;
  if (idx >= total) return;
  int ox = idx % 20; int t = idx / 20;
  int oy = t % 20;   t /= 20;
  int oc = t % 64;   int b = t / 64;
  const float* xb = x + (size_t)b * 17 * 84 * 84;
  const float* wp = w + (size_t)oc * 17 * 64;
  float acc = bias[oc];
  const int iy0 = oy * 4, ix0 = ox * 4;
  for (int c = 0; c < 17; ++c) {
    const float* xc = xb + (size_t)c * 84 * 84;
    const float* wc = wp + c * 64;
#pragma unroll
    for (int kh = 0; kh < 8; ++kh) {
      const float* xr = xc + (size_t)(iy0 + kh) * 84 + ix0;
      const float* wr = wc + kh * 8;
#pragma unroll
      for (int kw = 0; kw < 8; ++kw) acc = fmaf(xr[kw], wr[kw], acc);
    }
  }
  acc = acc > 0.f ? acc : 0.1f * acc;
  y[idx] = (_Float16)acc;
}

// ---------------------------------------------------------------------------
// im2col for conv2: A2[row=b*81+pos][k=ic*16+kh*4+kw] from y1h[b][ic][...]
// ---------------------------------------------------------------------------
__global__ void k_im2col(const _Float16* __restrict__ y1, _Float16* __restrict__ A2) {
  int idx = blockIdx.x * blockDim.x + threadIdx.x;
  const int total = 20736 * 1024;
  if (idx >= total) return;
  const int k   = idx & 1023;
  const int row = idx >> 10;
  const int ic = k >> 4, kh = (k >> 2) & 3, kw = k & 3;
  const int pos = row % 81, b = row / 81;
  const int oy = pos / 9, ox = pos % 9;
  A2[idx] = y1[(((size_t)b * 64 + ic) * 20 + (oy * 2 + kh)) * 20 + (ox * 2 + kw)];
}

// ---------------------------------------------------------------------------
// generic f32 -> f16 convert with trailing zero pad
// ---------------------------------------------------------------------------
__global__ void k_cvt_f32_f16(const float* __restrict__ src, _Float16* __restrict__ dst,
                              int n_src, int n_tot) {
  int i = blockIdx.x * blockDim.x + threadIdx.x;
  if (i >= n_tot) return;
  dst[i] = (i < n_src) ? (_Float16)src[i] : (_Float16)0.0f;
}

// ---------------------------------------------------------------------------
// coordinate augmentation columns 510/511 of e
// ---------------------------------------------------------------------------
__global__ void k_coords(float* __restrict__ e32, _Float16* __restrict__ eh) {
  int row = blockIdx.x * blockDim.x + threadIdx.x;
  if (row >= 20736) return;
  const int pos = row % 81;
  const float cy = (float)(pos / 9) / 9.0f;
  const float cx = (float)(pos % 9) / 9.0f;
  e32[(size_t)row * 512 + 510] = cy;
  e32[(size_t)row * 512 + 511] = cx;
  eh[(size_t)row * 512 + 510] = (_Float16)cy;
  eh[(size_t)row * 512 + 511] = (_Float16)cx;
}

// ---------------------------------------------------------------------------
// generic WMMA GEMM: C[M][N] = A[M][K](f16) * W[N][K]^T(f16), wave owns 16x64.
// Requires: M%16==0, N%64==0, K%64==0 (>=2 steps), lda/ldw multiples of 8.
// Branchless, register double-buffered (software pipelined) K loop.
// mode 0: plain f32 store
// mode 1: +bias, lrelu, skip cols >= ncut, optional dual f16 store
// mode 2: +residual (f32, stride N)
// ---------------------------------------------------------------------------
__global__ void k_gemm(const _Float16* __restrict__ A, int lda,
                       const _Float16* __restrict__ W, int ldw,
                       int M, int N, int K,
                       float* __restrict__ Cout, _Float16* __restrict__ Ch,
                       const float* __restrict__ bias,
                       const float* __restrict__ resid,
                       int mode, int ncut) {
  const int wid  = blockIdx.x * (blockDim.x >> 5) + (threadIdx.x >> 5);
  const int lane = threadIdx.x & (WAVE - 1);
  const int n64 = N >> 6;
  const int total = (M >> 4) * n64;
  if (wid >= total) return;                 // wave-uniform exit
  const int mt = wid / n64, nt = wid % n64;
  const int mbase = mt << 4, nbase0 = nt << 6;

  const int lm = lane & 15;
  const int ka = (lane >> 4) << 4;   // A-fragment K offset (0 or 16)
  const int kb4 = (lane >> 4) << 4;  // B-fragment K offset (0 or 16)
  const _Float16* arow = A + (size_t)(mbase + lm) * lda + ka;
  const _Float16* wrow = W + (size_t)(nbase0 + lm) * ldw + kb4;
  const size_t wstep = (size_t)ldw << 4;   // 16 rows of W

  __builtin_prefetch(arow, 0, 1);          // warm first A line (gfx1250 prefetch)

  v8f acc[4];
#pragma unroll
  for (int t = 0; t < 4; ++t)
#pragma unroll
    for (int r = 0; r < 8; ++r) acc[t][r] = 0.0f;

  // prologue: fragments for K-step 0
  v16h af = make_a(arow);
  v16h bf[4];
#pragma unroll
  for (int t = 0; t < 4; ++t) bf[t] = *(const v16h*)(wrow + (size_t)t * wstep);

  // steady state: load step kb while multiplying step kb-32
#pragma unroll 2
  for (int kb = 32; kb < K; kb += 32) {
    v16h af2 = make_a(arow + kb);
    v16h bf2[4];
#pragma unroll
    for (int t = 0; t < 4; ++t)
      bf2[t] = *(const v16h*)(wrow + (size_t)t * wstep + kb);
#pragma unroll
    for (int t = 0; t < 4; ++t) acc[t] = wmma16(af, bf[t], acc[t]);
    af = af2;
#pragma unroll
    for (int t = 0; t < 4; ++t) bf[t] = bf2[t];
  }
  // epilogue: last K-step
#pragma unroll
  for (int t = 0; t < 4; ++t) acc[t] = wmma16(af, bf[t], acc[t]);

  const int moff = (lane >> 4) << 3;
#pragma unroll
  for (int t = 0; t < 4; ++t) {
    const int n = nbase0 + (t << 4) + lm;
    if (mode == 1 && n >= ncut) continue;
#pragma unroll
    for (int r = 0; r < 8; ++r) {
      const int m = mbase + moff + r;
      float v = acc[t][r];
      if (mode == 1)      { v += bias[n]; v = v > 0.f ? v : 0.1f * v; }
      else if (mode == 2) { v += resid[(size_t)m * N + n]; }
      Cout[(size_t)m * N + n] = v;
      if (Ch) Ch[(size_t)m * N + n] = (_Float16)v;
    }
  }
}

// ---------------------------------------------------------------------------
// rowwise LayerNorm (biased var, eps=1e-6); optional relu; f16 and/or f32 out
// one 256-thread block per row; LDS tree reduction
// ---------------------------------------------------------------------------
__global__ void k_layernorm(const float* __restrict__ X,
                            const float* __restrict__ g, const float* __restrict__ b,
                            _Float16* __restrict__ Yh, float* __restrict__ Yf,
                            int D, int do_relu) {
  const int row = blockIdx.x;
  const int tid = threadIdx.x;
  const float* xr = X + (size_t)row * D;
  float s = 0.f, s2 = 0.f;
  for (int i = tid; i < D; i += blockDim.x) { float v = xr[i]; s += v; s2 += v * v; }
  __shared__ float sh0[256];
  __shared__ float sh1[256];
  sh0[tid] = s; sh1[tid] = s2;
  __syncthreads();
  for (int off = 128; off > 0; off >>= 1) {
    if (tid < off) { sh0[tid] += sh0[tid + off]; sh1[tid] += sh1[tid + off]; }
    __syncthreads();
  }
  const float mean = sh0[0] / (float)D;
  const float var  = sh1[0] / (float)D - mean * mean;
  const float inv  = rsqrtf(var + 1e-6f);
  for (int i = tid; i < D; i += blockDim.x) {
    float v = (xr[i] - mean) * inv * g[i] + b[i];
    if (do_relu) v = v > 0.f ? v : 0.f;
    if (Yh) Yh[(size_t)row * D + i] = (_Float16)v;
    if (Yf) Yf[(size_t)row * D + i] = v;
  }
}

// ---------------------------------------------------------------------------
// attention scores: per wave one 16x16 tile of S[b,h][n][m] = q.k^T / 16
// interior tiles (25/36) use branchless pipelined vector loads; edges predicated
// ---------------------------------------------------------------------------
__global__ void k_scores(const _Float16* __restrict__ qh,
                         const _Float16* __restrict__ kh,
                         float* __restrict__ S) {
  const int wid  = blockIdx.x * (blockDim.x >> 5) + (threadIdx.x >> 5);
  const int lane = threadIdx.x & (WAVE - 1);
  const int total = 256 * 4 * 36;
  if (wid >= total) return;
  const int tile = wid % 36;
  const int bh   = wid / 36;
  const int h = bh & 3, b = bh >> 2;
  const int mt = tile / 6, nt = tile % 6;
  const _Float16* Aq = qh + ((size_t)b * 81) * 1024 + h * 256;
  const _Float16* Wk = kh + ((size_t)b * 81) * 1024 + h * 256;
  v8f acc;
#pragma unroll
  for (int r = 0; r < 8; ++r) acc[r] = 0.0f;

  if (mt < 5 && nt < 5) {  // fully in-bounds tile: branchless + pipelined
    const int lm = lane & 15;
    const _Float16* arow = Aq + (size_t)((mt << 4) + lm) * 1024 + ((lane >> 4) << 4);
    const _Float16* wrow = Wk + (size_t)((nt << 4) + lm) * 1024 + ((lane >> 4) << 4);
    v16h af = make_a(arow);
    v16h bf = *(const v16h*)(wrow);
#pragma unroll 2
    for (int kb = 32; kb < 256; kb += 32) {
      v16h af2 = make_a(arow + kb);
      v16h bf2 = *(const v16h*)(wrow + kb);
      acc = wmma16(af, bf, acc);
      af = af2; bf = bf2;
    }
    acc = wmma16(af, bf, acc);
  } else {
    for (int kb = 0; kb < 256; kb += 32) {
      v16h af = load_a_pred(Aq, 1024, mt << 4, kb, lane, 81, 256);
      v16h bf = load_bT_pred(Wk, 1024, nt << 4, kb, lane, 81);
      acc = wmma16(af, bf, acc);
    }
  }

  const int ncol = (nt << 4) + (lane & 15);
  if (ncol < 81) {
    const int moff = (lane >> 4) << 3;
#pragma unroll
    for (int r = 0; r < 8; ++r) {
      const int m = (mt << 4) + moff + r;
      if (m < 81)
        S[((size_t)bh * 81 + m) * 81 + ncol] = acc[r] * 0.0625f;  // 1/sqrt(256)
    }
  }
}

// ---------------------------------------------------------------------------
// softmax over 81 keys; one 128-thread block per (b,h,n) row; f16 probs out
// ---------------------------------------------------------------------------
__global__ void k_softmax(const float* __restrict__ S, _Float16* __restrict__ P) {
  const int row = blockIdx.x;
  const int tid = threadIdx.x;
  const float* sr = S + (size_t)row * 81;
  __shared__ float sh[128];
  float mx = -1e30f;
  for (int i = tid; i < 81; i += 128) mx = fmaxf(mx, sr[i]);
  sh[tid] = mx; __syncthreads();
  for (int off = 64; off > 0; off >>= 1) {
    if (tid < off) sh[tid] = fmaxf(sh[tid], sh[tid + off]);
    __syncthreads();
  }
  mx = sh[0]; __syncthreads();
  float sum = 0.f;
  for (int i = tid; i < 81; i += 128) sum += __expf(sr[i] - mx);
  sh[tid] = sum; __syncthreads();
  for (int off = 64; off > 0; off >>= 1) {
    if (tid < off) sh[tid] += sh[tid + off];
    __syncthreads();
  }
  const float inv = 1.0f / sh[0];
  for (int i = tid; i < 81; i += 128)
    P[(size_t)row * 81 + i] = (_Float16)(__expf(sr[i] - mx) * inv);
}

// ---------------------------------------------------------------------------
// nv[b,n,h*256+d] = sum_m P[b,h][n][m] * v[b,m,h*256+d]; wave owns 16x16 tile
// ---------------------------------------------------------------------------
__global__ void k_attn_v(const _Float16* __restrict__ P,
                         const _Float16* __restrict__ vh,
                         _Float16* __restrict__ nvh) {
  const int wid  = blockIdx.x * (blockDim.x >> 5) + (threadIdx.x >> 5);
  const int lane = threadIdx.x & (WAVE - 1);
  const int total = 256 * 4 * 6 * 16;
  if (wid >= total) return;
  const int dnt = wid % 16;
  int rest = wid / 16;
  const int mt = rest % 6;
  const int bh = rest / 6;
  const int h = bh & 3, b = bh >> 2;
  const _Float16* Ap = P + (size_t)bh * 81 * 81;                   // [81][81]
  const _Float16* Bv = vh + ((size_t)b * 81) * 1024 + h * 256;     // [81][1024] slice
  v8f acc;
#pragma unroll
  for (int r = 0; r < 8; ++r) acc[r] = 0.0f;
  for (int kb = 0; kb < 81; kb += 32) {
    v16h af = load_a_pred(Ap, 81, mt << 4, kb, lane, 81, 81);
    v16h bf = load_b_direct(Bv, 1024, dnt << 4, kb, lane, 81);
    acc = wmma16(af, bf, acc);
  }
  const int d = (dnt << 4) + (lane & 15);
  const int moff = (lane >> 4) << 3;
#pragma unroll
  for (int r = 0; r < 8; ++r) {
    const int m = (mt << 4) + moff + r;
    if (m < 81)
      nvh[((size_t)b * 81 + m) * 1024 + h * 256 + d] = (_Float16)acc[r];
  }
}

// ---------------------------------------------------------------------------
// max over the 81 entities (post relu-LN), dual f32/f16 out [256][512]
// ---------------------------------------------------------------------------
__global__ void k_maxent(const float* __restrict__ res,
                         float* __restrict__ fin, _Float16* __restrict__ finh) {
  int idx = blockIdx.x * blockDim.x + threadIdx.x;
  if (idx >= 256 * 512) return;
  const int b = idx >> 9, c = idx & 511;
  const float* p = res + ((size_t)b * 81) * 512 + c;
  float mx = p[0];
  for (int n = 1; n < 81; ++n) mx = fmaxf(mx, p[(size_t)n * 512]);
  fin[idx] = mx;
  finh[idx] = (_Float16)mx;
}

// ---------------------------------------------------------------------------
// host-side orchestration
// ---------------------------------------------------------------------------
extern "C" void kernel_launch(void* const* d_in, const int* in_sizes, int n_in,
                              void* d_out, int out_size, void* d_ws, size_t ws_size,
                              hipStream_t stream) {
  (void)in_sizes; (void)n_in; (void)out_size; (void)ws_size;
  const float* x       = (const float*)d_in[0];
  const float* conv1_w = (const float*)d_in[1];
  const float* conv1_b = (const float*)d_in[2];
  const float* conv2_w = (const float*)d_in[3];
  const float* conv2_b = (const float*)d_in[4];
  const float* wq      = (const float*)d_in[5];
  const float* wk      = (const float*)d_in[6];
  const float* wv      = (const float*)d_in[7];
  const float* ln_e_g  = (const float*)d_in[8];
  const float* ln_e_b  = (const float*)d_in[9];
  const float* wo      = (const float*)d_in[10];
  const float* ln_g    = (const float*)d_in[11];
  const float* ln_b    = (const float*)d_in[12];
  const float* map_w   = (const float*)d_in[13];
  const float* map_b   = (const float*)d_in[14];
  float* out = (float*)d_out;

  const int ROWS = 256 * 81;  // 20736

  // workspace carve-up (256B aligned)
  char* base = (char*)d_ws;
  size_t off = 0;
  auto carve = [&](size_t bytes) -> char* {
    char* p = base + off;
    off += (bytes + 255) & ~(size_t)255;
    return p;
  };
  _Float16* y1h  = (_Float16*)carve((size_t)256 * 64 * 20 * 20 * 2);
  _Float16* A2   = (_Float16*)carve((size_t)ROWS * 1024 * 2);
  _Float16* w2h  = (_Float16*)carve((size_t)512 * 1024 * 2);
  _Float16* wqh  = (_Float16*)carve((size_t)1024 * 512 * 2);
  _Float16* wkh  = (_Float16*)carve((size_t)1024 * 512 * 2);
  _Float16* wvh  = (_Float16*)carve((size_t)1024 * 512 * 2);
  _Float16* woh  = (_Float16*)carve((size_t)512 * 1024 * 2);
  _Float16* maph = (_Float16*)carve((size_t)256 * 512 * 2);
  float*    e32  = (float*)   carve((size_t)ROWS * 512 * 4);
  _Float16* eh   = (_Float16*)carve((size_t)ROWS * 512 * 2);
  float*    raw  = (float*)   carve((size_t)ROWS * 1024 * 4);  // reused 4x
  _Float16* qh   = (_Float16*)carve((size_t)ROWS * 1024 * 2);
  _Float16* kh   = (_Float16*)carve((size_t)ROWS * 1024 * 2);
  _Float16* vh   = (_Float16*)carve((size_t)ROWS * 1024 * 2);
  float*    S    = (float*)   carve((size_t)256 * 4 * 81 * 81 * 4);
  _Float16* Ph   = (_Float16*)carve((size_t)256 * 4 * 81 * 81 * 2);
  _Float16* nvh  = (_Float16*)carve((size_t)ROWS * 1024 * 2);
  float*    res  = (float*)   carve((size_t)ROWS * 512 * 4);
  float*    fin  = (float*)   carve((size_t)256 * 512 * 4);
  _Float16* finh = (_Float16*)carve((size_t)256 * 512 * 2);

  auto cdiv = [](int a, int b) { return (a + b - 1) / b; };

  // weight conversions (f32 -> f16), conv2 padded 510 -> 512 rows
  k_cvt_f32_f16<<<cdiv(512 * 1024, 256), 256, 0, stream>>>(conv2_w, w2h, 510 * 1024, 512 * 1024);
  k_cvt_f32_f16<<<cdiv(1024 * 512, 256), 256, 0, stream>>>(wq, wqh, 1024 * 512, 1024 * 512);
  k_cvt_f32_f16<<<cdiv(1024 * 512, 256), 256, 0, stream>>>(wk, wkh, 1024 * 512, 1024 * 512);
  k_cvt_f32_f16<<<cdiv(1024 * 512, 256), 256, 0, stream>>>(wv, wvh, 1024 * 512, 1024 * 512);
  k_cvt_f32_f16<<<cdiv(512 * 1024, 256), 256, 0, stream>>>(wo, woh, 512 * 1024, 512 * 1024);
  k_cvt_f32_f16<<<cdiv(256 * 512, 256), 256, 0, stream>>>(map_w, maph, 256 * 512, 256 * 512);

  // conv1 + im2col
  k_conv1<<<cdiv(256 * 64 * 20 * 20, 256), 256, 0, stream>>>(x, conv1_w, conv1_b, y1h);
  k_im2col<<<cdiv(ROWS * 1024, 256), 256, 0, stream>>>(y1h, A2);

  // conv2 as WMMA GEMM: [20736,1024] x [512,1024]^T, bias+lrelu, dual store
  {
    int waves = (ROWS / 16) * (512 / 64);
    k_gemm<<<cdiv(waves, 8), 256, 0, stream>>>(A2, 1024, w2h, 1024, ROWS, 512, 1024,
                                               e32, eh, conv2_b, nullptr, 1, 510);
  }
  k_coords<<<cdiv(ROWS, 256), 256, 0, stream>>>(e32, eh);

  // Q/K/V projections + LayerNorm(1024)
  {
    int waves = (ROWS / 16) * (1024 / 64);
    int blocks = cdiv(waves, 8);
    k_gemm<<<blocks, 256, 0, stream>>>(eh, 512, wqh, 512, ROWS, 1024, 512,
                                       raw, nullptr, nullptr, nullptr, 0, 0);
    k_layernorm<<<ROWS, 256, 0, stream>>>(raw, ln_e_g, ln_e_b, qh, nullptr, 1024, 0);
    k_gemm<<<blocks, 256, 0, stream>>>(eh, 512, wkh, 512, ROWS, 1024, 512,
                                       raw, nullptr, nullptr, nullptr, 0, 0);
    k_layernorm<<<ROWS, 256, 0, stream>>>(raw, ln_e_g, ln_e_b, kh, nullptr, 1024, 0);
    k_gemm<<<blocks, 256, 0, stream>>>(eh, 512, wvh, 512, ROWS, 1024, 512,
                                       raw, nullptr, nullptr, nullptr, 0, 0);
    k_layernorm<<<ROWS, 256, 0, stream>>>(raw, ln_e_g, ln_e_b, vh, nullptr, 1024, 0);
  }

  // attention
  k_scores<<<cdiv(256 * 4 * 36, 8), 256, 0, stream>>>(qh, kh, S);
  k_softmax<<<256 * 4 * 81, 128, 0, stream>>>(S, Ph);
  k_attn_v<<<cdiv(256 * 4 * 6 * 16, 8), 256, 0, stream>>>(Ph, vh, nvh);

  // output projection + residual, LayerNorm(512)+relu, entity max
  {
    int waves = (ROWS / 16) * (512 / 64);
    k_gemm<<<cdiv(waves, 8), 256, 0, stream>>>(nvh, 1024, woh, 1024, ROWS, 512, 1024,
                                               raw, nullptr, nullptr, e32, 2, 0);
  }
  k_layernorm<<<ROWS, 256, 0, stream>>>(raw, ln_g, ln_b, nullptr, res, 512, 1);
  k_maxent<<<cdiv(256 * 512, 256), 256, 0, stream>>>(res, fin, finh);

  // final mapping: [256,512] x [256,512]^T + bias, lrelu -> d_out [256,256]
  {
    int waves = (256 / 16) * (256 / 64);
    k_gemm<<<cdiv(waves, 8), 256, 0, stream>>>(finh, 512, maph, 512, 256, 256, 512,
                                               out, nullptr, map_b, nullptr, 1, 256);
  }
}
